// GlobalAveragePoolingWithAttention_54116587929972
// MI455X (gfx1250) — matrix operand order repaired and checked by reading, not compiled, not run on hardware
//
#include <hip/hip_runtime.h>

typedef __attribute__((ext_vector_type(16))) _Float16 v16h;
typedef __attribute__((ext_vector_type(8)))  float    v8f;
typedef __attribute__((ext_vector_type(8)))  unsigned int v8u;

#define B_  64
#define C_  256
#define N_  4096    // 64*64 tokens
#define K_  2048    // keep_num

// ---------------------------------------------------------------------------
// Kernel 1: per-token ABS score.  score[b][n] = sum_c |x[b,c,n]|
// Each thread owns 4 consecutive tokens -> float4 coalesced streaming reads.
// ---------------------------------------------------------------------------
__global__ __launch_bounds__(256) void score_kernel(const float* __restrict__ x,
                                                    float* __restrict__ score) {
  int idx = blockIdx.x * blockDim.x + threadIdx.x;   // 0 .. B*N/4-1
  int n0  = (idx << 2) & (N_ - 1);
  int b   = (idx << 2) >> 12;
  const float* p = x + (size_t)b * C_ * N_ + n0;
  float s0 = 0.f, s1 = 0.f, s2 = 0.f, s3 = 0.f;
  #pragma unroll 4
  for (int c = 0; c < C_; ++c) {
    float4 v = *(const float4*)(p + (size_t)c * N_);
    s0 += fabsf(v.x); s1 += fabsf(v.y); s2 += fabsf(v.z); s3 += fabsf(v.w);
  }
  float4 r; r.x = s0; r.y = s1; r.z = s2; r.w = s3;
  *(float4*)(score + (size_t)b * N_ + n0) = r;
}

// ---------------------------------------------------------------------------
// Kernel 2: per-batch exact top-K selection -> f16 {0,1} mask.
// One block (1024 threads = 32 waves) per batch. Bitonic sort of the 4096
// scores in LDS to find the K-th largest value; ties resolved by lowest
// index (matches jax.lax.top_k). Integer LDS atomics only (deterministic).
// ---------------------------------------------------------------------------
__global__ __launch_bounds__(1024) void select_kernel(const float* __restrict__ score,
                                                      _Float16* __restrict__ maskh) {
  __shared__ float sv[N_];   // sorted copy
  __shared__ float su[N_];   // original order
  __shared__ int   cntG;
  const int b   = blockIdx.x;
  const int tid = threadIdx.x;
  const float* sc = score + (size_t)b * N_;

  for (int i = tid; i < N_; i += 1024) { float v = sc[i]; sv[i] = v; su[i] = v; }
  __syncthreads();

  // ascending bitonic sort of sv[]
  for (int ksz = 2; ksz <= N_; ksz <<= 1) {
    for (int j = ksz >> 1; j > 0; j >>= 1) {
      for (int i = tid; i < N_; i += 1024) {
        int ixj = i ^ j;
        if (ixj > i) {
          bool up = ((i & ksz) == 0);
          float a = sv[i], c = sv[ixj];
          if ((a > c) == up) { sv[i] = c; sv[ixj] = a; }
        }
      }
      __syncthreads();
    }
  }

  const float thr = sv[N_ - K_];          // K-th largest score
  if (tid == 0) cntG = 0;
  __syncthreads();
  int lc = 0;
  for (int i = tid; i < N_; i += 1024) lc += (su[i] > thr) ? 1 : 0;
  atomicAdd(&cntG, lc);
  __syncthreads();
  const int need = K_ - cntG;             // ties to take (lowest indices first)

  for (int i = tid; i < N_; i += 1024) {
    float s = su[i];
    _Float16 mv = (_Float16)0.f;
    if (s > thr) {
      mv = (_Float16)1.f;
    } else if (s == thr) {
      int r = 0;
      for (int m = 0; m < i; ++m) r += (su[m] == thr) ? 1 : 0;
      if (r < need) mv = (_Float16)1.f;
    }
    maskh[(size_t)b * N_ + i] = mv;
  }
}

// ---------------------------------------------------------------------------
// Kernel 3: masked channel means as WMMA GEMV.
//   partial[b][c][ks] = sum over tokens in K-slice ks of mask[n]*x[b,c,n]
// One wave per 16-channel tile; A = 16x32 f32->f16 tile (contiguous k:
// lanes 0-15 hold K0..15, lanes 16-31 hold K16..31),
// B = mask replicated over 16 columns (lanes 0-15: K0..15, lanes 16-31:
// K16..31), f32 accumulator. Column N=0 of D (lanes 0 & 16) is the result.
// ---------------------------------------------------------------------------
__global__ __launch_bounds__(256) void wmma_pool_kernel(const float* __restrict__ x,
                                                        const _Float16* __restrict__ maskh,
                                                        float* __restrict__ partial) {
  __shared__ __align__(32) _Float16 smask[1024];
  const int b     = blockIdx.x;        // batch
  const int chalf = blockIdx.y;        // channel half: tiles 0-7 / 8-15
  const int ks    = blockIdx.z;        // K-slice of 1024 tokens
  const int tid   = threadIdx.x;

  // stage this slice's mask into LDS (broadcast-read later by all waves)
  for (int i = tid; i < 1024; i += 256)
    smask[i] = maskh[(size_t)b * N_ + ks * 1024 + i];
  __syncthreads();

  const int wave = tid >> 5;
  const int lane = tid & 31;
  const int half = lane >> 4;          // which K-half of the wave
  const int m    = lane & 15;          // row (channel) within tile
  const int ct   = chalf * 8 + wave;   // channel tile 0..15

  const float* rowp = x + (size_t)(b * C_ + ct * 16 + m) * N_ + ks * 1024;

  v8f acc = {};
  for (int ch = 0; ch < 32; ++ch) {
    const int k0 = ch * 32;
    // A operand: this lane's 16 K-values of its channel row, f32 -> f16
    const float* p = rowp + k0 + half * 16;
    float4 f0 = *(const float4*)(p);
    float4 f1 = *(const float4*)(p + 4);
    float4 f2 = *(const float4*)(p + 8);
    float4 f3 = *(const float4*)(p + 12);
    v16h a;
    a[0]  = (_Float16)f0.x; a[1]  = (_Float16)f0.y; a[2]  = (_Float16)f0.z; a[3]  = (_Float16)f0.w;
    a[4]  = (_Float16)f1.x; a[5]  = (_Float16)f1.y; a[6]  = (_Float16)f1.z; a[7]  = (_Float16)f1.w;
    a[8]  = (_Float16)f2.x; a[9]  = (_Float16)f2.y; a[10] = (_Float16)f2.z; a[11] = (_Float16)f2.w;
    a[12] = (_Float16)f3.x; a[13] = (_Float16)f3.y; a[14] = (_Float16)f3.z; a[15] = (_Float16)f3.w;

    // B operand: mask values, identical for all 16 columns.
    // lanes 0-15 need K = k0..k0+15, lanes 16-31 need K = k0+16..k0+31.
    const int kb = k0 + half * 16;                    // multiple of 16 -> 32B aligned
    v8u bw = *(const v8u*)(smask + kb);
    v16h bm = __builtin_bit_cast(v16h, bw);

    acc = __builtin_amdgcn_wmma_f32_16x16x32_f16(
        /*neg_a=*/false, a, /*neg_b=*/false, bm,
        /*c_mod=*/(short)0, acc, /*reuse_a=*/false, /*reuse_b=*/false);
  }

  // D layout: VGPR j holds (M=j, N=lane) for lanes 0-15 and (M=8+j, N=lane-16)
  // for lanes 16-31.  Column N=0 lives in lanes 0 and 16.
  if (m == 0) {
    const int mbase = half * 8;
    #pragma unroll
    for (int j = 0; j < 8; ++j)
      partial[(size_t)(b * C_ + ct * 16 + mbase + j) * 4 + ks] = acc[j];
  }
}

// ---------------------------------------------------------------------------
// Kernel 4: reduce the 4 K-slice partials and scale by 1/K.
// ---------------------------------------------------------------------------
__global__ __launch_bounds__(256) void finalize_kernel(const float* __restrict__ partial,
                                                       float* __restrict__ out) {
  int i = blockIdx.x * blockDim.x + threadIdx.x;
  if (i < B_ * C_) {
    const float* p = partial + (size_t)i * 4;
    out[i] = (p[0] + p[1] + p[2] + p[3]) * (1.0f / (float)K_);
  }
}

// ---------------------------------------------------------------------------
extern "C" void kernel_launch(void* const* d_in, const int* in_sizes, int n_in,
                              void* d_out, int out_size, void* d_ws, size_t ws_size,
                              hipStream_t stream) {
  const float* x  = (const float*)d_in[0];
  float* out      = (float*)d_out;

  char* ws            = (char*)d_ws;
  float*    score     = (float*)ws;                                     // 1 MB
  _Float16* maskh     = (_Float16*)(ws + (size_t)B_ * N_ * sizeof(float));      // 512 KB
  float*    partial   = (float*)(ws + (size_t)B_ * N_ * sizeof(float)
                                    + (size_t)B_ * N_ * sizeof(_Float16));      // 256 KB

  score_kernel   <<<dim3((B_ * N_ / 4) / 256), dim3(256),  0, stream>>>(x, score);
  select_kernel  <<<dim3(B_),                  dim3(1024), 0, stream>>>(score, maskh);
  wmma_pool_kernel<<<dim3(B_, 2, 4),           dim3(256),  0, stream>>>(x, maskh, partial);
  finalize_kernel<<<dim3((B_ * C_ + 255) / 256), dim3(256), 0, stream>>>(partial, out);
}
